// PAMLayerReal_90993177133458
// MI455X (gfx1250) — matrix operand order repaired and checked by reading, not compiled, not run on hardware
//
#include <hip/hip_runtime.h>
#include <hip/hip_bf16.h>
#include <stdint.h>

// Problem constants (B, T, D, H, d) = (2, 2048, 1024, 16, 64)
#define BB 2
#define TT 2048
#define DD 1024
#define HH 16
#define dd_ 64
#define INNER 1024
#define N3 3072
#define BT 4096
#define SCALE 0.125f

typedef __attribute__((ext_vector_type(16))) __bf16 v16bf;
typedef __attribute__((ext_vector_type(8)))  float  v8f;

// ---------------------------------------------------------------------------
// WMMA fragment loaders (wave32, 16-bit data, contiguous k = 16*half + s in A and B)
__device__ inline v16bf load_a_frag(const __bf16* row, int half) {
  v16bf a;
  const uint32_t* r32 = reinterpret_cast<const uint32_t*>(row);
  uint32_t* ap = reinterpret_cast<uint32_t*>(&a);
  int o = 8 * half;
#pragma unroll
  for (int i = 0; i < 4; ++i) ap[i] = r32[o + i];         // K = 16*half + 2i
#pragma unroll
  for (int i = 0; i < 4; ++i) ap[4 + i] = r32[o + 4 + i]; // K = 16*half + 8 + 2i
  return a;
}

__device__ inline v16bf load_b_frag(const __bf16* row, int half) {
  v16bf b;
  const uint32_t* r32 = reinterpret_cast<const uint32_t*>(row + 16 * half);
  uint32_t* bp = reinterpret_cast<uint32_t*>(&b);
#pragma unroll
  for (int i = 0; i < 8; ++i) bp[i] = r32[i];
  return b;
}

__device__ inline v8f wmma_bf16(v16bf a, v16bf b, v8f c) {
  return __builtin_amdgcn_wmma_f32_16x16x32_bf16(false, a, false, b,
                                                 (short)0, c, false, false);
}

// ---------------------------------------------------------------------------
// CDNA5 async global->LDS DMA (ASYNCcnt-tracked), 16 bytes/lane.
// Generic LDS pointers carry the LDS byte address in their low 32 bits.
__device__ inline void async_load_b128(void* lds_ptr, const void* gptr) {
  uint32_t lds_addr = (uint32_t)(uintptr_t)lds_ptr;
  asm volatile("global_load_async_to_lds_b128 %0, %1, off"
               :: "v"(lds_addr), "v"(gptr)
               : "memory");
}
__device__ inline void wait_async() {
  asm volatile("s_wait_asynccnt 0" ::: "memory");
}

// ---------------------------------------------------------------------------
// fp32 -> bf16 conversion
__global__ void cvt_f32_bf16(const float* __restrict__ in,
                             __bf16* __restrict__ out, int n) {
  int i = blockIdx.x * blockDim.x + threadIdx.x;
  if (i < n) out[i] = (__bf16)in[i];
}

// fp32 KxN -> bf16 NxK transpose (weights; done once, feeds async GEMM loads)
__global__ void cvt_t_f32_bf16(const float* __restrict__ in,
                               __bf16* __restrict__ out, int K, int N) {
  int gid = blockIdx.x * blockDim.x + threadIdx.x;
  if (gid >= K * N) return;
  int n = gid / K, k = gid - n * K;
  out[(size_t)n * K + k] = (__bf16)in[(size_t)k * N + n];
}

// ---------------------------------------------------------------------------
// bf16 GEMM: C(MxN fp32) = A(MxK) * Bt(NxK)^T. M%128==0, N%128==0, K%32==0.
// 256 threads = 8 waves (2x4), block tile 128x128, wave tile 64x32,
// double-buffered async global->LDS staging overlapped with WMMA.
__global__ __launch_bounds__(256) void gemm_bf16(
    const __bf16* __restrict__ A, const __bf16* __restrict__ Bt,
    float* __restrict__ C, int M, int N, int K) {
  __shared__ __attribute__((aligned(16))) __bf16 As[2][128][40]; // 80B rows
  __shared__ __attribute__((aligned(16))) __bf16 Bs[2][128][40]; // [n][k]
  const int tid = threadIdx.x;
  const int wave = tid >> 5, lane = tid & 31;
  const int half = lane >> 4, l16 = lane & 15;
  const int wm = (wave >> 2) * 64, wn = (wave & 3) * 32;
  const int m0 = blockIdx.y * 128, n0 = blockIdx.x * 128;
  const int nk = K >> 5;

  auto stage = [&](int kt, int buf) {
    int k0 = kt * 32;
    // 128 rows x 32 bf16 = 512 x 16B chunks per operand; 2 chunks/thread.
#pragma unroll
    for (int it = 0; it < 2; ++it) {
      int idx = tid + it * 256;
      int row = idx >> 2, ch = (idx & 3) * 8;
      async_load_b128(&As[buf][row][ch], A + (size_t)(m0 + row) * K + k0 + ch);
      async_load_b128(&Bs[buf][row][ch], Bt + (size_t)(n0 + row) * K + k0 + ch);
    }
  };

  v8f acc[4][2] = {};
  stage(0, 0);
  wait_async();
  __syncthreads();
  for (int kt = 0; kt < nk; ++kt) {
    int cur = kt & 1;
    if (kt + 1 < nk) stage(kt + 1, cur ^ 1);  // DMA next tile during compute
    v16bf a[4];
#pragma unroll
    for (int mi = 0; mi < 4; ++mi)
      a[mi] = load_a_frag(&As[cur][wm + mi * 16 + l16][0], half);
#pragma unroll
    for (int ni = 0; ni < 2; ++ni) {
      v16bf b = load_b_frag(&Bs[cur][wn + ni * 16 + l16][0], half);
#pragma unroll
      for (int mi = 0; mi < 4; ++mi)
        acc[mi][ni] = wmma_bf16(a[mi], b, acc[mi][ni]);
    }
    wait_async();
    __syncthreads();
  }
#pragma unroll
  for (int mi = 0; mi < 4; ++mi)
#pragma unroll
    for (int ni = 0; ni < 2; ++ni)
#pragma unroll
      for (int r = 0; r < 8; ++r) {
        int m = m0 + wm + mi * 16 + r + 8 * half;
        int n = n0 + wn + ni * 16 + l16;
        C[(size_t)m * N + n] = acc[mi][ni][r];
      }
}

// ---------------------------------------------------------------------------
// dt = softplus(x @ W_dt + b_dt); gamma = exp(-dt). One thread per (b*T+t, h).
__global__ void dt_gamma_kernel(const float* __restrict__ x,
                                const float* __restrict__ W_dt,
                                const float* __restrict__ b_dt,
                                float* __restrict__ gamma_bth) {
  int gid = blockIdx.x * blockDim.x + threadIdx.x;
  if (gid >= BT * HH) return;
  int h = gid & (HH - 1), bt = gid >> 4;
  const float* xr = x + (size_t)bt * DD;
  float acc = 0.f;
  for (int k = 0; k < DD; ++k) acc += xr[k] * W_dt[k * HH + h];
  float z = acc + b_dt[h];
  float sp = fmaxf(z, 0.f) + log1pf(expf(-fabsf(z)));  // softplus, stable
  gamma_bth[gid] = expf(-sp);
}

// ---------------------------------------------------------------------------
// Repack: RoPE on Q/K, p=sigmoid gate, V*(1-p), per-(b,h,t) log(gamma_scalar).
__global__ __launch_bounds__(64) void repack_kernel(
    const float* __restrict__ qkv, const float* __restrict__ gsp,
    const float* __restrict__ gamma_bth, const float* __restrict__ b_gsp,
    const float* __restrict__ inv_freq, __bf16* __restrict__ Qo,
    __bf16* __restrict__ Ko, __bf16* __restrict__ Vo,
    float* __restrict__ log_gamma) {
  int blk = blockIdx.x;           // (b*T + t)*H + h
  int h = blk & (HH - 1);
  int bt = blk >> 4;
  int t = bt & (TT - 1);
  int b = bt >> 11;
  int dd = threadIdx.x;
  size_t rowq = (size_t)bt * N3;
  int col = h * dd_ + dd;

  float q = qkv[rowq + col];
  float k = qkv[rowq + INNER + col];
  float v = qkv[rowq + 2 * INNER + col];
  int parc = h * dd_ + (dd ^ 1);
  float qp = qkv[rowq + parc];
  float kp = qkv[rowq + INNER + parc];

  float ang = (float)t * inv_freq[dd >> 1];
  float c = cosf(ang), s = sinf(ang);
  float qr, kr;
  if ((dd & 1) == 0) { qr = q * c - qp * s; kr = k * c - kp * s; }
  else               { qr = qp * s + q * c; kr = kp * s + k * c; }

  float g = gsp[(size_t)bt * INNER + col] + b_gsp[col];
  float p = 1.f / (1.f + expf(-g));
  float gam = gamma_bth[(size_t)bt * HH + h];
  float ge = gam * (1.f - p) + p;

  size_t bh = (size_t)(b * HH + h);
  size_t o = (bh * TT + t) * dd_ + dd;
  Qo[o] = (__bf16)(qr * SCALE);
  Ko[o] = (__bf16)kr;
  Vo[o] = (__bf16)(v * (1.f - p));

  __shared__ float red[64];
  red[dd] = ge; __syncthreads();
  if (dd < 32) red[dd] += red[dd + 32]; __syncthreads();
  if (dd < 16) red[dd] += red[dd + 16]; __syncthreads();
  if (dd < 8)  red[dd] += red[dd + 8];  __syncthreads();
  if (dd < 4)  red[dd] += red[dd + 4];  __syncthreads();
  if (dd < 2)  red[dd] += red[dd + 2];  __syncthreads();
  if (dd == 0) {
    float gs = (red[0] + red[1]) * (1.f / 64.f);
    gs = fminf(fmaxf(gs, 1e-6f), 1.f - 1e-6f);
    log_gamma[bh * TT + t] = logf(gs);
  }
}

// ---------------------------------------------------------------------------
// Per-(b,h) serial cumsum of log_gamma over T (32 sequences x 2048).
__global__ void cumsum_kernel(const float* __restrict__ lg,
                              float* __restrict__ lgc) {
  int bh = blockIdx.x;
  if (threadIdx.x == 0) {
    float acc = 0.f;
    const float* in = lg + (size_t)bh * TT;
    float* out = lgc + (size_t)bh * TT;
    for (int t = 0; t < TT; ++t) { acc += in[t]; out[t] = acc; }
  }
}

// ---------------------------------------------------------------------------
// Flash-style decay attention. Grid: (T/64 t-tiles, B*H heads), 128 threads.
// Q/K tiles staged via async DMA; S = Q K^T (WMMA), decay+causal applied in
// fp32 accumulators, A re-swizzled through per-wave LDS, Y += A V (WMMA).
__global__ __launch_bounds__(128) void attn_kernel(
    const __bf16* __restrict__ Q, const __bf16* __restrict__ K,
    const __bf16* __restrict__ V, const float* __restrict__ lgc,
    __bf16* __restrict__ Yo) {
  __shared__ __attribute__((aligned(16))) __bf16 Qs[64][72];  // 144B rows
  __shared__ __attribute__((aligned(16))) __bf16 Ks[64][72];
  __shared__ __attribute__((aligned(16))) __bf16 Vt[64][66];  // [d][s]
  __shared__ __attribute__((aligned(16))) __bf16 Ss[4][16][66];
  __shared__ float lt[64];
  __shared__ float ls[64];

  const int bh = blockIdx.y;
  const int t0 = blockIdx.x * 64;
  const int tid = threadIdx.x;
  const int wave = tid >> 5, lane = tid & 31;
  const int half = lane >> 4, l16 = lane & 15;
  const int trow = wave * 16;

  const __bf16* Qh = Q + (size_t)bh * TT * dd_;
  const __bf16* Kh = K + (size_t)bh * TT * dd_;
  const __bf16* Vh = V + (size_t)bh * TT * dd_;
  const float* lg = lgc + (size_t)bh * TT;

  // Q tile: 64 rows x 64 bf16 = 256 x 16B chunks, async DMA (2 chunks/thread)
#pragma unroll
  for (int it = 0; it < 2; ++it) {
    int idx = tid + it * 128;
    int r = idx >> 2, ch = (idx & 3) * 16;  // ch in bf16 elements
    async_load_b128(&Qs[r][ch], Qh + (size_t)(t0 + r) * dd_ + ch);
  }
  if (tid < 64) lt[tid] = lg[t0 + tid];

  v8f y[4] = {};
  for (int s0 = 0; s0 <= t0; s0 += 64) {
    __syncthreads();  // previous iter done reading Ks/Vt/ls
#pragma unroll
    for (int it = 0; it < 2; ++it) {
      int idx = tid + it * 128;
      int r = idx >> 2, ch = (idx & 3) * 16;
      async_load_b128(&Ks[r][ch], Kh + (size_t)(s0 + r) * dd_ + ch);
    }
    for (int idx = tid; idx < 64 * 64; idx += 128) {  // V transposed (scalar)
      int s = idx >> 6, dcol = idx & 63;
      Vt[dcol][s] = Vh[(size_t)(s0 + s) * dd_ + dcol];
    }
    if (tid < 64) ls[tid] = lg[s0 + tid];
    wait_async();     // Q (iter 0) + K DMA complete
    __syncthreads();

    // S = Q * K^T  (wave: 16t x 64s, K-dim = d = 64 -> 2 WMMA steps)
    v8f sacc[4] = {};
#pragma unroll
    for (int kk = 0; kk < 2; ++kk) {
      v16bf a = load_a_frag(&Qs[trow + l16][kk * 32], half);
#pragma unroll
      for (int ni = 0; ni < 4; ++ni) {
        v16bf b = load_b_frag(&Ks[ni * 16 + l16][kk * 32], half);
        sacc[ni] = wmma_bf16(a, b, sacc[ni]);
      }
    }
    // decay * causal, emit bf16 A tile into per-wave LDS scratch
#pragma unroll
    for (int ni = 0; ni < 4; ++ni)
#pragma unroll
      for (int r = 0; r < 8; ++r) {
        int mt = trow + r + 8 * half;
        int ns = ni * 16 + l16;
        float dec = 0.f;
        if (s0 + ns <= t0 + mt) {
          float ld = lt[mt] - ls[ns];
          ld = fminf(fmaxf(ld, -20.f), 0.f);
          dec = expf(ld);
        }
        Ss[wave][r + 8 * half][ni * 16 + l16] = (__bf16)(sacc[ni][r] * dec);
      }
    // Y += A * V  (same-wave LDS RAW: DS pipe is in-order per wave)
#pragma unroll
    for (int kk = 0; kk < 2; ++kk) {
      v16bf a = load_a_frag(&Ss[wave][l16][kk * 32], half);
#pragma unroll
      for (int ni = 0; ni < 4; ++ni) {
        v16bf b = load_b_frag(&Vt[ni * 16 + l16][kk * 32], half);
        y[ni] = wmma_bf16(a, b, y[ni]);
      }
    }
  }

  const int b = bh >> 4, h = bh & (HH - 1);
#pragma unroll
  for (int ni = 0; ni < 4; ++ni)
#pragma unroll
    for (int r = 0; r < 8; ++r) {
      int t = t0 + trow + r + 8 * half;
      int dcol = ni * 16 + l16;
      Yo[(size_t)(b * TT + t) * INNER + h * dd_ + dcol] = (__bf16)y[ni][r];
    }
}

// ---------------------------------------------------------------------------
extern "C" void kernel_launch(void* const* d_in, const int* in_sizes, int n_in,
                              void* d_out, int out_size, void* d_ws,
                              size_t ws_size, hipStream_t stream) {
  const float* x        = (const float*)d_in[0];
  const float* W_qkv    = (const float*)d_in[1];
  const float* W_out    = (const float*)d_in[2];
  const float* W_dt     = (const float*)d_in[3];
  const float* b_dt     = (const float*)d_in[4];
  const float* W_gsp    = (const float*)d_in[5];
  const float* b_gsp    = (const float*)d_in[6];
  const float* inv_freq = (const float*)d_in[7];
  float* out = (float*)d_out;

  char* ws = (char*)d_ws;
  size_t off = 0;
  auto take = [&](size_t bytes) {
    char* p = ws + off;
    off += (bytes + 255) & ~(size_t)255;
    return p;
  };
  __bf16* x_bf    = (__bf16*)take((size_t)BT * DD * 2);
  __bf16* wqkv_t  = (__bf16*)take((size_t)DD * N3 * 2);     // N3 x DD
  __bf16* wgsp_t  = (__bf16*)take((size_t)DD * INNER * 2);  // INNER x DD
  __bf16* wout_t  = (__bf16*)take((size_t)INNER * DD * 2);  // DD x INNER
  float*  qkv_f   = (float*)take((size_t)BT * N3 * 4);
  float*  gsp_f   = (float*)take((size_t)BT * INNER * 4);
  float*  gamma   = (float*)take((size_t)BT * HH * 4);
  __bf16* Qbf     = (__bf16*)take((size_t)BB * HH * TT * dd_ * 2);
  __bf16* Kbf     = (__bf16*)take((size_t)BB * HH * TT * dd_ * 2);
  __bf16* Vbf     = (__bf16*)take((size_t)BB * HH * TT * dd_ * 2);
  float*  lgm     = (float*)take((size_t)BB * HH * TT * 4);
  float*  lgc     = (float*)take((size_t)BB * HH * TT * 4);
  __bf16* Ybf     = (__bf16*)take((size_t)BT * INNER * 2);

  // 1) bf16 conversions (weights transposed once so GEMM B-tiles are
  //    row-contiguous for async B128 DMA)
  {
    int n;
    n = BT * DD;    cvt_f32_bf16<<<(n + 255) / 256, 256, 0, stream>>>(x, x_bf, n);
    n = DD * N3;    cvt_t_f32_bf16<<<(n + 255) / 256, 256, 0, stream>>>(W_qkv, wqkv_t, DD, N3);
    n = DD * INNER; cvt_t_f32_bf16<<<(n + 255) / 256, 256, 0, stream>>>(W_gsp, wgsp_t, DD, INNER);
    n = INNER * DD; cvt_t_f32_bf16<<<(n + 255) / 256, 256, 0, stream>>>(W_out, wout_t, INNER, DD);
  }
  // 2) qkv = x @ W_qkv ; gsp = x @ W_gsp  (WMMA bf16, fp32 accumulate)
  gemm_bf16<<<dim3(N3 / 128, BT / 128), 256, 0, stream>>>(x_bf, wqkv_t, qkv_f, BT, N3, DD);
  gemm_bf16<<<dim3(INNER / 128, BT / 128), 256, 0, stream>>>(x_bf, wgsp_t, gsp_f, BT, INNER, DD);
  // 3) gamma from dt path
  dt_gamma_kernel<<<(BT * HH) / 256, 256, 0, stream>>>(x, W_dt, b_dt, gamma);
  // 4) RoPE + gate + repack + log_gamma
  repack_kernel<<<BT * HH, 64, 0, stream>>>(qkv_f, gsp_f, gamma, b_gsp, inv_freq,
                                            Qbf, Kbf, Vbf, lgm);
  // 5) cumulative log decay per head
  cumsum_kernel<<<BB * HH, 32, 0, stream>>>(lgm, lgc);
  // 6) tiled decay attention (WMMA, no T x T materialization)
  attn_kernel<<<dim3(TT / 64, BB * HH), 128, 0, stream>>>(Qbf, Kbf, Vbf, lgc, Ybf);
  // 7) out = Y @ W_out
  gemm_bf16<<<dim3(DD / 128, BT / 128), 256, 0, stream>>>(Ybf, wout_t, out, BT, DD, INNER);
}
